// DynamicAttention7_40827959116541
// MI455X (gfx1250) — matrix operand order repaired and checked by reading, not compiled, not run on hardware
//
#include <hip/hip_runtime.h>
#include <hip/hip_bf16.h>

// ---------------------------------------------------------------------------
// Problem constants
// ---------------------------------------------------------------------------
#define BB   16
#define LL   128
#define SS   256
#define TT   256
#define QD   512
#define FD   1024
#define HD   256
#define OD   256
#define SP1  257          // S+1 (with pad row)
#define SPAD 272          // padded to 17*16 for WMMA tiling
#define MROW 273          // LDS row stride for score matrix (bank-friendly)

typedef __attribute__((ext_vector_type(16))) __bf16 v16bf;
typedef __attribute__((ext_vector_type(8)))  __bf16 v8bf;
typedef __attribute__((ext_vector_type(8)))  float  v8f;

__device__ inline v16bf pack16(v8bf lo, v8bf hi) {
  v16bf r;
#pragma unroll
  for (int i = 0; i < 8; ++i) { r[i] = lo[i]; r[8 + i] = hi[i]; }
  return r;
}

__device__ inline v8f wmma_bf16(v16bf a, v16bf b, v8f c) {
  // D = A(16x32 bf16) * B(32x16 bf16) + C(16x16 f32)
  return __builtin_amdgcn_wmma_f32_16x16x32_bf16(false, a, false, b,
                                                 (short)0, c, false, false);
}

// ---------------------------------------------------------------------------
// fp32 -> bf16 convert (grid-stride)
// ---------------------------------------------------------------------------
__global__ void cvt_bf16_kernel(const float* __restrict__ in,
                                __bf16* __restrict__ out, long n) {
  long i = (long)blockIdx.x * blockDim.x + threadIdx.x;
  long stride = (long)gridDim.x * blockDim.x;
  for (; i < n; i += stride) out[i] = (__bf16)in[i];
}

// fp32 [K][N] -> bf16 [N][K] (transpose for WMMA B-fragment loads)
__global__ void cvt_bf16_T_kernel(const float* __restrict__ in,
                                  __bf16* __restrict__ out, int K, int N) {
  long total = (long)K * N;
  long i = (long)blockIdx.x * blockDim.x + threadIdx.x;
  long stride = (long)gridDim.x * blockDim.x;
  for (; i < total; i += stride) {
    int n = (int)(i / K);
    int k = (int)(i - (long)n * K);
    out[i] = (__bf16)in[(long)k * N + n];
  }
}

// ---------------------------------------------------------------------------
// Generic bf16 WMMA projection GEMM:  C = A[M x K] * Bt[N=256 x K]^T + bias
// One 16x16 tile per wave, 8 waves per block. Writes fp32 and/or bf16 output
// with per-batch row strides (keys get a pad-row hole at s=256..271).
// ---------------------------------------------------------------------------
__global__ __launch_bounds__(256) void proj_gemm_kernel(
    const __bf16* __restrict__ A, const __bf16* __restrict__ Bt,
    const float* __restrict__ bias, int Mrows, int K, int rows_per_batch,
    float* __restrict__ outF32, long f32_bstride,
    __bf16* __restrict__ outBF16, long bf16_bstride) {
  int gid = blockIdx.x * 8 + (threadIdx.x >> 5);
  int lane = threadIdx.x & 31;
  int tile_m = gid >> 4;      // 16 n-tiles (N = 256)
  int tile_n = gid & 15;
  if (tile_m * 16 >= Mrows) return;
  int lm = lane & 15;
  int lh = lane >> 4;

  const __bf16* arow = A + (size_t)(tile_m * 16 + lm) * K;
  const __bf16* brow = Bt + (size_t)(tile_n * 16 + lm) * K + lh * 8;

  v8f acc = {};
  for (int k0 = 0; k0 < K; k0 += 32) {
    v8bf alo = *(const v8bf*)(arow + k0 + lh * 8);
    v8bf ahi = *(const v8bf*)(arow + k0 + 16 + lh * 8);
    v8bf blo = *(const v8bf*)(brow + k0);
    v8bf bhi = *(const v8bf*)(brow + k0 + 16);
    acc = wmma_bf16(pack16(alo, ahi), pack16(blo, bhi), acc);
  }

  int col = tile_n * 16 + lm;
  float bv = bias[col];
#pragma unroll
  for (int i = 0; i < 8; ++i) {
    int row = tile_m * 16 + lh * 8 + i;
    int bidx = row / rows_per_batch;
    int s = row - bidx * rows_per_batch;
    float v = acc[i] + bv;
    if (outF32)  outF32[(size_t)bidx * f32_bstride + (size_t)s * 256 + col] = v;
    if (outBF16) outBF16[(size_t)bidx * bf16_bstride + (size_t)s * 256 + col] = (__bf16)v;
  }
}

// ---------------------------------------------------------------------------
// Fill pad row (s=256) with alpha, zero rows 257..271 of bf16 key buffers.
// ---------------------------------------------------------------------------
__global__ void pad_fill_kernel(float* skf, float* tkf, __bf16* skb,
                                __bf16* tkb, const float* alphaPtr) {
  int b = blockIdx.x;
  int h = threadIdx.x;
  float a = alphaPtr[0];
  skf[((size_t)b * SP1 + 256) * HD + h] = a;
  tkf[((size_t)b * SP1 + 256) * HD + h] = a;
  skb[((size_t)b * SPAD + 256) * HD + h] = (__bf16)a;
  tkb[((size_t)b * SPAD + 256) * HD + h] = (__bf16)a;
  for (int r = 257; r < SPAD; ++r) {
    skb[((size_t)b * SPAD + r) * HD + h] = (__bf16)0.0f;
    tkb[((size_t)b * SPAD + r) * HD + h] = (__bf16)0.0f;
  }
}

// ---------------------------------------------------------------------------
// Fused trilinear attention: one workgroup per (b,l).
//  - WMMA score GEMM: M[s,t] = (q_l ⊙ s_key_s) · t_key_t / 16, kept in LDS
//  - softmax over 257x257 (max, exp-in-place, row sums ws, col sums wt, Z)
//  - s_ctx = (ws/Z) @ s_key, t_ctx = (wt/Z) @ t_key  (coalesced GEMVs)
//  - out = relu(concat(query, s_ctx, t_ctx) @ Wo + bo)
// Dynamic LDS: 272*273*4 (M) + 512 (q) + 1088 (ws) + 1088 (wt) + 1024 (red)
// ---------------------------------------------------------------------------
#define SMEM_M    0
#define SMEM_Q    (SPAD * MROW * 4)          // 297024
#define SMEM_WS   (SMEM_Q + 512)             // 297536
#define SMEM_WT   (SMEM_WS + 1088)           // 298624
#define SMEM_RED  (SMEM_WT + 1088)           // 299712
#define SMEM_TOT  (SMEM_RED + 1024)          // 300736

__global__ __launch_bounds__(256) void attn_kernel(
    const __bf16* __restrict__ qbf,   // [B*L][256]
    const __bf16* __restrict__ skb,   // [B][272][256]
    const __bf16* __restrict__ tkb,   // [B][272][256]
    const float* __restrict__ skf,    // [B][257][256]
    const float* __restrict__ tkf,    // [B][257][256]
    const float* __restrict__ query,  // [B][L][512]
    const float* __restrict__ Wo,     // [1024][256]
    const float* __restrict__ bo,     // [256]
    float* __restrict__ out) {        // [B][L][256]
  extern __shared__ char smem[];
  float* Msh  = (float*)(smem + SMEM_M);
  __bf16* qsh = (__bf16*)(smem + SMEM_Q);
  float* wssh = (float*)(smem + SMEM_WS);
  float* wtsh = (float*)(smem + SMEM_WT);
  float* red  = (float*)(smem + SMEM_RED);

  const int tid = threadIdx.x;
  const int bl = blockIdx.x;            // b*128 + l
  const int b = bl >> 7;
  const int wave = tid >> 5;
  const int lane = tid & 31;
  const int lm = lane & 15;
  const int lh = lane >> 4;

  // ---- stage q_l (bf16) in LDS ----
  qsh[tid] = qbf[(size_t)bl * HD + tid];
  __syncthreads();

  // ---- Phase 1: score GEMM into LDS (17x17 tiles of 16x16) ----
  const __bf16* skbase = skb + (size_t)b * SPAD * HD;
  const __bf16* tkbase = tkb + (size_t)b * SPAD * HD;
  for (int si = wave; si < 17; si += 8) {
    // Build q-scaled A fragments for this si strip (reused across 17 ti tiles)
    v16bf afr[8];
    {
      const __bf16* arow = skbase + (size_t)(si * 16 + lm) * HD;
#pragma unroll
      for (int kc = 0; kc < 8; ++kc) {
        int k0 = kc * 32 + lh * 8;
        v8bf lo = *(const v8bf*)(arow + k0);
        v8bf hi = *(const v8bf*)(arow + k0 + 16);
        v8bf ql = *(const v8bf*)(qsh + k0);
        v8bf qh = *(const v8bf*)(qsh + k0 + 16);
#pragma unroll
        for (int i = 0; i < 8; ++i) {
          afr[kc][i]     = (__bf16)((float)lo[i] * (float)ql[i]);
          afr[kc][8 + i] = (__bf16)((float)hi[i] * (float)qh[i]);
        }
      }
    }
    for (int ti = 0; ti < 17; ++ti) {
      v8f acc = {};
      const __bf16* brow = tkbase + (size_t)(ti * 16 + lm) * HD + lh * 8;
#pragma unroll
      for (int kc = 0; kc < 8; ++kc) {
        v8bf blo = *(const v8bf*)(brow + kc * 32);
        v8bf bhi = *(const v8bf*)(brow + kc * 32 + 16);
        acc = wmma_bf16(afr[kc], pack16(blo, bhi), acc);
      }
      int r0 = si * 16 + lh * 8;
      int c  = ti * 16 + lm;
#pragma unroll
      for (int i = 0; i < 8; ++i)
        Msh[(r0 + i) * MROW + c] = acc[i] * 0.0625f;  // 1/sqrt(H)
    }
  }
  __syncthreads();

  // ---- Phase 2: global max over valid 257x257 ----
  float lmax = -3.4e38f;
  for (int r = tid; r < SP1; r += 256)
    for (int t = 0; t < SP1; ++t)
      lmax = fmaxf(lmax, Msh[r * MROW + t]);
  red[tid] = lmax;
  __syncthreads();
  for (int s = 128; s > 0; s >>= 1) {
    if (tid < s) red[tid] = fmaxf(red[tid], red[tid + s]);
    __syncthreads();
  }
  float mx = red[0];
  __syncthreads();

  // ---- Phase 3: exp in place + row sums (ws) ----
  for (int r = tid; r < SP1; r += 256) {
    float acc = 0.0f;
    for (int t = 0; t < SP1; ++t) {
      float p = __expf(Msh[r * MROW + t] - mx);
      Msh[r * MROW + t] = p;
      acc += p;
    }
    wssh[r] = acc;
  }
  __syncthreads();

  // ---- Phase 4: col sums (wt) + Z ----
  for (int c = tid; c < SP1; c += 256) {
    float acc = 0.0f;
    for (int r = 0; r < SP1; ++r) acc += Msh[r * MROW + c];
    wtsh[c] = acc;
  }
  float lz = 0.0f;
  for (int r = tid; r < SP1; r += 256) lz += wssh[r];
  red[tid] = lz;
  __syncthreads();
  for (int s = 128; s > 0; s >>= 1) {
    if (tid < s) red[tid] += red[tid + s];
    __syncthreads();
  }
  float inv = 1.0f / red[0];
  __syncthreads();

  // ---- Phase 5: context GEMVs (coalesced over h = tid) ----
  float sc = 0.0f, tc = 0.0f;
  {
    const float* sk = skf + (size_t)b * SP1 * HD + tid;
    const float* tk = tkf + (size_t)b * SP1 * HD + tid;
    for (int s = 0; s < SP1; ++s) sc += wssh[s] * sk[(size_t)s * HD];
    for (int t = 0; t < SP1; ++t) tc += wtsh[t] * tk[(size_t)t * HD];
  }
  sc *= inv;
  tc *= inv;
  __syncthreads();  // done reading Msh; reuse as x buffer

  // ---- Phase 6: x = concat(query, s_ctx, t_ctx); out = relu(x@Wo + bo) ----
  float* xsh = Msh;
  xsh[tid]       = query[(size_t)bl * QD + tid];
  xsh[tid + 256] = query[(size_t)bl * QD + 256 + tid];
  xsh[512 + tid] = sc;
  xsh[768 + tid] = tc;
  __syncthreads();

  float acc = bo[tid];
  const float* w = Wo + tid;
  for (int j = 0; j < 2 * HD + QD; ++j) acc += xsh[j] * w[(size_t)j * OD];
  out[(size_t)bl * OD + tid] = fmaxf(acc, 0.0f);
}

// ---------------------------------------------------------------------------
// Host launcher
// ---------------------------------------------------------------------------
extern "C" void kernel_launch(void* const* d_in, const int* in_sizes, int n_in,
                              void* d_out, int out_size, void* d_ws,
                              size_t ws_size, hipStream_t stream) {
  (void)in_sizes; (void)n_in; (void)out_size; (void)ws_size;
  const float* query = (const float*)d_in[0];
  const float* src   = (const float*)d_in[1];
  const float* trg   = (const float*)d_in[2];
  const float* Wq    = (const float*)d_in[3];
  const float* bq    = (const float*)d_in[4];
  const float* Ws    = (const float*)d_in[5];
  const float* bs    = (const float*)d_in[6];
  const float* alpha = (const float*)d_in[7];
  const float* Wo    = (const float*)d_in[8];
  const float* bo    = (const float*)d_in[9];
  float* out = (float*)d_out;

  // Workspace layout (bytes, all 512B-aligned). Total ~33.6 MB.
  char* ws = (char*)d_ws;
  size_t o_qbf  = 0;                                    // [2048][256] bf16
  size_t o_skf  = o_qbf  + (size_t)BB * LL * HD * 2;    // [16][257][256] f32
  size_t o_tkf  = o_skf  + (size_t)BB * SP1 * HD * 4;
  size_t o_skb  = o_tkf  + (size_t)BB * SP1 * HD * 4;   // [16][272][256] bf16
  size_t o_tkb  = o_skb  + (size_t)BB * SPAD * HD * 2;
  size_t o_qryb = o_tkb  + (size_t)BB * SPAD * HD * 2;  // [2048][512] bf16
  size_t o_srcb = o_qryb + (size_t)BB * LL * QD * 2;    // [4096][1024] bf16
  size_t o_trgb = o_srcb + (size_t)BB * SS * FD * 2;
  size_t o_wqt  = o_trgb + (size_t)BB * TT * FD * 2;    // [256][512] bf16
  size_t o_wst  = o_wqt  + (size_t)HD * QD * 2;         // [256][1024] bf16

  __bf16* qbf  = (__bf16*)(ws + o_qbf);
  float*  skf  = (float*)(ws + o_skf);
  float*  tkf  = (float*)(ws + o_tkf);
  __bf16* skb  = (__bf16*)(ws + o_skb);
  __bf16* tkb  = (__bf16*)(ws + o_tkb);
  __bf16* qryb = (__bf16*)(ws + o_qryb);
  __bf16* srcb = (__bf16*)(ws + o_srcb);
  __bf16* trgb = (__bf16*)(ws + o_trgb);
  __bf16* wqt  = (__bf16*)(ws + o_wqt);
  __bf16* wst  = (__bf16*)(ws + o_wst);

  // 1) fp32 -> bf16 conversions (and weight transposes)
  cvt_bf16_kernel<<<1024, 256, 0, stream>>>(query, qryb, (long)BB * LL * QD);
  cvt_bf16_kernel<<<2048, 256, 0, stream>>>(src, srcb, (long)BB * SS * FD);
  cvt_bf16_kernel<<<2048, 256, 0, stream>>>(trg, trgb, (long)BB * TT * FD);
  cvt_bf16_T_kernel<<<512, 256, 0, stream>>>(Wq, wqt, QD, HD);
  cvt_bf16_T_kernel<<<1024, 256, 0, stream>>>(Ws, wst, FD, HD);

  // 2) projections (WMMA GEMMs)
  // q = query @ Wq + bq  -> bf16 only
  proj_gemm_kernel<<<(BB * LL / 16) * 16 / 8, 256, 0, stream>>>(
      qryb, wqt, bq, BB * LL, QD, LL,
      nullptr, 0, qbf, (long)LL * HD);
  // s_key = src @ Ws + bs -> fp32 (257-row layout) + bf16 (272-row layout)
  proj_gemm_kernel<<<(BB * SS / 16) * 16 / 8, 256, 0, stream>>>(
      srcb, wst, bs, BB * SS, FD, SS,
      skf, (long)SP1 * HD, skb, (long)SPAD * HD);
  // t_key = trg @ Ws + bs
  proj_gemm_kernel<<<(BB * TT / 16) * 16 / 8, 256, 0, stream>>>(
      trgb, wst, bs, BB * TT, FD, TT,
      tkf, (long)SP1 * HD, tkb, (long)SPAD * HD);

  // 3) pad rows (alpha at s=256, zeros at 257..271 in bf16 buffers)
  pad_fill_kernel<<<BB, 256, 0, stream>>>(skf, tkf, skb, tkb, alpha);

  // 4) fused trilinear attention + softmax + ctx + output projection
  attn_kernel<<<BB * LL, 256, SMEM_TOT, stream>>>(
      qbf, skb, tkb, skf, tkf, query, Wo, bo, out);
}
